// SanctaTransformer_11656541241720
// MI455X (gfx1250) — matrix operand order repaired and checked by reading, not compiled, not run on hardware
//
#include <hip/hip_runtime.h>
#include <math.h>

typedef __attribute__((ext_vector_type(16))) _Float16 v16h;
typedef __attribute__((ext_vector_type(8)))  _Float16 h8;
typedef __attribute__((ext_vector_type(8)))  float    v8f;

#define SS   64   // seq len
#define DD   32   // d_model
#define DFF_ 64
#define NL_  2
#define HD_  8
#define EPS_ 1e-5f

static __device__ __forceinline__ v8f wmma16(v16h a, v16h b, v8f c) {
  return __builtin_amdgcn_wmma_f32_16x16x32_f16(false, a, false, b, (short)0, c, false, false);
}

static __device__ __forceinline__ v16h cat8(h8 lo, h8 hi) {
  return __builtin_shufflevector(lo, hi, 0, 1, 2, 3, 4, 5, 6, 7,
                                         8, 9, 10, 11, 12, 13, 14, 15);
}

// A fragment: 16x32 f16 tile, rows row0..row0+15 from row-major [row][stride] halves.
// lanes 0-15 -> M=lane, elems 0..15 = K 0..15;
//      lanes 16-31 -> same rows, K 16..31.
// All call sites have 16B-aligned addresses (stride 32/64 halves, k offsets % 8 == 0).
static __device__ __forceinline__ v16h load_A(const _Float16* base, int stride, int row0, int k0) {
  const int lane = threadIdx.x & 31;
  const int m    = lane & 15;
  const int sub  = (lane >> 4) << 4;               // 0 or 16
  const _Float16* p = base + (row0 + m) * stride + k0 + sub;
  return cat8(*(const h8*)p, *(const h8*)(p + 8));
}

// B fragment: 32x16 f16 tile, B[k][n] = w[n0+n][k0+k] from [N][K] row-major weights.
// lanes 0-15: N=lane, K 0..15 contiguous; lanes 16-31: N=lane-16, K 16..31.
static __device__ __forceinline__ v16h load_B(const _Float16* w, int stride, int n0, int k0) {
  const int lane = threadIdx.x & 31;
  const int n    = n0 + (lane & 15);
  const int klo  = k0 + ((lane >> 4) << 4);        // 0 or 16 within chunk
  const _Float16* p = w + n * stride + klo;
  return cat8(*(const h8*)p, *(const h8*)(p + 8));
}

// B for Q@K^T, head h: B[k=d][n=j] = K[j0+j][d] if d in [8h,8h+8) else 0.
static __device__ __forceinline__ v16h load_B_scores(const _Float16* khp, int j0, int h) {
  const int lane = threadIdx.x & 31;
  const int j = j0 + (lane & 15);
  v16h b = {};
  if ((h >> 1) == (lane >> 4)) {                   // head's 8 K-rows fall in this lane group
    const h8 z = {};
    const h8 val = *(const h8*)(khp + j * DD + h * HD_);
    b = (h & 1) ? cat8(z, val) : cat8(val, z);
  }
  return b;
}

// B for attn@V, head h: B[k=tok][n] = V[k0+tok][ncol] (vt transposed [dim][tok]),
// masked to ncol in [8h,8h+8); tile column base is (h>>1)*16.
static __device__ __forceinline__ v16h load_B_av(const _Float16* vtp, int h, int k0) {
  const int lane = threadIdx.x & 31;
  const int ncol = ((h >> 1) << 4) + (lane & 15);
  v16h b = {};
  if ((ncol >> 3) == h) {
    const int klo = k0 + ((lane >> 4) << 4);
    const _Float16* p = vtp + ncol * SS + klo;
    b = cat8(*(const h8*)p, *(const h8*)(p + 8));
  }
  return b;
}

// ---------------- prep: sinusoidal PE + fp32 -> f16 weight images ----------------
__global__ void sancta_prep_kernel(const float* __restrict__ wqkv, const float* __restrict__ wo,
                                   const float* __restrict__ w1, const float* __restrict__ w2,
                                   float* __restrict__ pe, _Float16* __restrict__ w16) {
  const int total = SS * DD + 16384;
  for (int i = blockIdx.x * blockDim.x + threadIdx.x; i < total; i += gridDim.x * blockDim.x) {
    if (i < SS * DD) {
      int t = i >> 5, d = i & 31;
      float div = __expf(-(float)(2 * (d >> 1)) * (logf(10000.0f) / (float)DD));
      float ang = (float)t * div;
      pe[i] = (d & 1) ? cosf(ang) : sinf(ang);
    } else {
      int j = i - SS * DD;
      float v;
      if (j < NL_ * 96 * DD)                 v = wqkv[j];
      else if (j < NL_ * 96 * DD + NL_ * DD * DD)
        v = wo[j - NL_ * 96 * DD];
      else if (j < NL_ * 96 * DD + NL_ * DD * DD + NL_ * DFF_ * DD)
        v = w1[j - NL_ * 96 * DD - NL_ * DD * DD];
      else
        v = w2[j - NL_ * 96 * DD - NL_ * DD * DD - NL_ * DFF_ * DD];
      w16[j] = (_Float16)v;
    }
  }
}

// ---------------- fused transformer: 1 block = 1 sequence, 4 waves ----------------
__global__ void __launch_bounds__(128) sancta_tfm_kernel(
    const int* __restrict__ token_ids, const float* __restrict__ embed_W,
    const float* __restrict__ bqkv, const float* __restrict__ bo,
    const float* __restrict__ ln1_g, const float* __restrict__ ln1_b,
    const float* __restrict__ ln2_g, const float* __restrict__ ln2_b,
    const float* __restrict__ b1, const float* __restrict__ b2,
    const float* __restrict__ lnf_g, const float* __restrict__ lnf_b,
    const float* __restrict__ pe, const _Float16* __restrict__ w16,
    float* __restrict__ out) {
  __shared__ float    xf[SS][DD];      // activations fp32
  __shared__ _Float16 xh[SS][DD];      // activations f16 (A operand)
  __shared__ _Float16 qh[SS][DD];
  __shared__ _Float16 kh[SS][DD];
  __shared__ _Float16 vt[DD][SS];      // V transposed [dim][token]
  __shared__ _Float16 sc[4][16][SS];   // per-wave softmax probs
  __shared__ _Float16 oh[SS][DD];      // attention output f16
  __shared__ _Float16 hh[SS][DFF_];    // FFN hidden f16 (reused as 8KB staging in phase 0)

  const int tid  = threadIdx.x;
  const int wave = tid >> 5;
  const int lane = tid & 31;
  const int seq  = blockIdx.x;
  const int row0 = wave * 16;
  const int nlo  = lane & 15;                      // C-fragment column
  const int mb   = row0 + ((lane >> 4) << 3);      // C-fragment row base

  if (tid == 0) {                                  // warm L2/WGP$ for weights
    __builtin_prefetch(w16, 0, 3);
    __builtin_prefetch((const char*)w16 + 16384, 0, 3);
  }

  // Phase 0: async-DMA the gathered embedding rows into LDS (ASYNCcnt path),
  // then combine with the positional encoding: x = 2*embed[tok] + pe.
  {
    float* stage = (float*)&hh[0][0];              // 64*32 f32 = 8KB staging
    for (int chunk = tid; chunk < SS * 8; chunk += 128) {   // 512 x 16B chunks
      int t = chunk >> 3, c = chunk & 7;
      int tok = token_ids[seq * SS + t];
      unsigned voff   = (unsigned)(tok * (DD * 4) + c * 16);
      unsigned ldsoff = (unsigned)(uintptr_t)&stage[t * DD + c * 4];
      asm volatile("global_load_async_to_lds_b128 %0, %1, %2"
                   :: "v"(ldsoff), "v"(voff), "s"(embed_W)
                   : "memory");
    }
    asm volatile("s_wait_asynccnt 0" ::: "memory");
    __syncthreads();
    for (int idx = tid; idx < SS * DD; idx += 128) {
      float v = 2.0f * stage[idx] + pe[idx];
      xf[idx >> 5][idx & 31] = v;
      xh[idx >> 5][idx & 31] = (_Float16)v;
    }
  }

  const _Float16* wqkv16 = w16;                    // [NL][96][32]
  const _Float16* wo16   = wqkv16 + NL_ * 96 * DD; // [NL][32][32]
  const _Float16* w116   = wo16 + NL_ * DD * DD;   // [NL][64][32]
  const _Float16* w216   = w116 + NL_ * DFF_ * DD; // [NL][32][64]

  for (int l = 0; l < NL_; ++l) {
    __syncthreads();   // phase0 visible / previous layer's K,V readers done

    // ---- Phase 1: QKV = x @ Wqkv^T + b ----
    {
      const _Float16* Wq = wqkv16 + l * 96 * DD;
      v16h ax = load_A(&xh[0][0], DD, row0, 0);
#pragma unroll
      for (int nt = 0; nt < 6; ++nt) {
        v8f c = {};
        c = wmma16(ax, load_B(Wq, DD, nt * 16, 0), c);
        const int n0 = nt * 16;
        const float bias = bqkv[l * 96 + n0 + nlo];
        if (n0 < 32) {
#pragma unroll
          for (int r = 0; r < 8; ++r) qh[mb + r][n0 + nlo] = (_Float16)(c[r] + bias);
        } else if (n0 < 64) {
#pragma unroll
          for (int r = 0; r < 8; ++r) kh[mb + r][n0 - 32 + nlo] = (_Float16)(c[r] + bias);
        } else {
#pragma unroll
          for (int r = 0; r < 8; ++r) vt[n0 - 64 + nlo][mb + r] = (_Float16)(c[r] + bias);
        }
      }
    }
    __syncthreads();   // all K,V in LDS

    // ---- Phase 2: attention (per wave: its 16 query rows) ----
    v8f oc0 = {}, oc1 = {};
    {
      const v16h aq = load_A(&qh[0][0], DD, row0, 0);  // full 32-dim Q rows; head mask in B
      _Float16* scW = &sc[wave][0][0];
      const float iscale = 0.35355339059327373f;        // 1/sqrt(8)
      const int rl = (lane >> 4) << 3;
#pragma unroll
      for (int h = 0; h < 4; ++h) {
        v8f cjt[4];
#pragma unroll
        for (int jt = 0; jt < 4; ++jt) {
          v8f z = {};
          cjt[jt] = wmma16(aq, load_B_scores(&kh[0][0], jt * 16, h), z);
        }
        float mx[8], sm[8];
#pragma unroll
        for (int r = 0; r < 8; ++r)
          mx[r] = fmaxf(fmaxf(cjt[0][r], cjt[1][r]), fmaxf(cjt[2][r], cjt[3][r]));
#pragma unroll
        for (int msk = 1; msk <= 8; msk <<= 1)
#pragma unroll
          for (int r = 0; r < 8; ++r) mx[r] = fmaxf(mx[r], __shfl_xor(mx[r], msk, 32));
#pragma unroll
        for (int r = 0; r < 8; ++r) sm[r] = 0.0f;
#pragma unroll
        for (int jt = 0; jt < 4; ++jt)
#pragma unroll
          for (int r = 0; r < 8; ++r) {
            float e = __expf((cjt[jt][r] - mx[r]) * iscale);
            cjt[jt][r] = e;
            sm[r] += e;
          }
#pragma unroll
        for (int msk = 1; msk <= 8; msk <<= 1)
#pragma unroll
          for (int r = 0; r < 8; ++r) sm[r] += __shfl_xor(sm[r], msk, 32);
#pragma unroll
        for (int r = 0; r < 8; ++r) sm[r] = 1.0f / sm[r];
#pragma unroll
        for (int jt = 0; jt < 4; ++jt)
#pragma unroll
          for (int r = 0; r < 8; ++r)
            scW[(rl + r) * SS + jt * 16 + nlo] = (_Float16)(cjt[jt][r] * sm[r]);
        // attn @ V (K = 64 tokens -> two chunks), accumulate into output dims tile
        v16h a0 = load_A(scW, SS, 0, 0);
        v16h a1 = load_A(scW, SS, 0, 32);
        if (h < 2) {
          oc0 = wmma16(a0, load_B_av(&vt[0][0], h, 0), oc0);
          oc0 = wmma16(a1, load_B_av(&vt[0][0], h, 32), oc0);
        } else {
          oc1 = wmma16(a0, load_B_av(&vt[0][0], h, 0), oc1);
          oc1 = wmma16(a1, load_B_av(&vt[0][0], h, 32), oc1);
        }
      }
    }

    // ---- Phase 3: output projection + residual + LN1 ----
    {
#pragma unroll
      for (int r = 0; r < 8; ++r) {
        oh[mb + r][nlo]      = (_Float16)oc0[r];
        oh[mb + r][16 + nlo] = (_Float16)oc1[r];
      }
      const _Float16* Wo = wo16 + l * DD * DD;
      v16h ao = load_A(&oh[0][0], DD, row0, 0);
      v8f p0 = {}, p1 = {};
      p0 = wmma16(ao, load_B(Wo, DD, 0, 0), p0);
      p1 = wmma16(ao, load_B(Wo, DD, 16, 0), p1);
      const float g0 = ln1_g[l * DD + nlo],      be0 = ln1_b[l * DD + nlo];
      const float g1 = ln1_g[l * DD + 16 + nlo], be1 = ln1_b[l * DD + 16 + nlo];
      const float bo0 = bo[l * DD + nlo], bo1 = bo[l * DD + 16 + nlo];
      float v0[8], v1[8], su[8], sq[8];
#pragma unroll
      for (int r = 0; r < 8; ++r) {
        v0[r] = xf[mb + r][nlo] + p0[r] + bo0;
        v1[r] = xf[mb + r][16 + nlo] + p1[r] + bo1;
        su[r] = v0[r] + v1[r];
        sq[r] = v0[r] * v0[r] + v1[r] * v1[r];
      }
#pragma unroll
      for (int msk = 1; msk <= 8; msk <<= 1)
#pragma unroll
        for (int r = 0; r < 8; ++r) {
          su[r] += __shfl_xor(su[r], msk, 32);
          sq[r] += __shfl_xor(sq[r], msk, 32);
        }
#pragma unroll
      for (int r = 0; r < 8; ++r) {
        float mean = su[r] * (1.0f / DD);
        float var  = sq[r] * (1.0f / DD) - mean * mean;
        float rstd = rsqrtf(var + EPS_);
        float y0 = (v0[r] - mean) * rstd * g0 + be0;
        float y1 = (v1[r] - mean) * rstd * g1 + be1;
        xf[mb + r][nlo] = y0;      xh[mb + r][nlo] = (_Float16)y0;
        xf[mb + r][16 + nlo] = y1; xh[mb + r][16 + nlo] = (_Float16)y1;
      }
    }

    // ---- Phase 4: FFN + residual + LN2 ----
    {
      const _Float16* W1 = w116 + l * DFF_ * DD;
      const _Float16* W2 = w216 + l * DD * DFF_;
      v16h ax2 = load_A(&xh[0][0], DD, row0, 0);
#pragma unroll
      for (int nt = 0; nt < 4; ++nt) {
        v8f c = {};
        c = wmma16(ax2, load_B(W1, DD, nt * 16, 0), c);
        const float bb = b1[l * DFF_ + nt * 16 + nlo];
#pragma unroll
        for (int r = 0; r < 8; ++r) hh[mb + r][nt * 16 + nlo] = (_Float16)fmaxf(c[r] + bb, 0.0f);
      }
      v16h ah0 = load_A(&hh[0][0], DFF_, row0, 0);
      v16h ah1 = load_A(&hh[0][0], DFF_, row0, 32);
      v8f f0 = {}, f1 = {};
      f0 = wmma16(ah0, load_B(W2, DFF_, 0, 0), f0);
      f0 = wmma16(ah1, load_B(W2, DFF_, 0, 32), f0);
      f1 = wmma16(ah0, load_B(W2, DFF_, 16, 0), f1);
      f1 = wmma16(ah1, load_B(W2, DFF_, 16, 32), f1);
      const float g0 = ln2_g[l * DD + nlo],      be0 = ln2_b[l * DD + nlo];
      const float g1 = ln2_g[l * DD + 16 + nlo], be1 = ln2_b[l * DD + 16 + nlo];
      const float bb0 = b2[l * DD + nlo], bb1 = b2[l * DD + 16 + nlo];
      float v0[8], v1[8], su[8], sq[8];
#pragma unroll
      for (int r = 0; r < 8; ++r) {
        v0[r] = xf[mb + r][nlo] + f0[r] + bb0;
        v1[r] = xf[mb + r][16 + nlo] + f1[r] + bb1;
        su[r] = v0[r] + v1[r];
        sq[r] = v0[r] * v0[r] + v1[r] * v1[r];
      }
#pragma unroll
      for (int msk = 1; msk <= 8; msk <<= 1)
#pragma unroll
        for (int r = 0; r < 8; ++r) {
          su[r] += __shfl_xor(su[r], msk, 32);
          sq[r] += __shfl_xor(sq[r], msk, 32);
        }
#pragma unroll
      for (int r = 0; r < 8; ++r) {
        float mean = su[r] * (1.0f / DD);
        float var  = sq[r] * (1.0f / DD) - mean * mean;
        float rstd = rsqrtf(var + EPS_);
        float y0 = (v0[r] - mean) * rstd * g0 + be0;
        float y1 = (v1[r] - mean) * rstd * g1 + be1;
        xf[mb + r][nlo] = y0;      xh[mb + r][nlo] = (_Float16)y0;
        xf[mb + r][16 + nlo] = y1; xh[mb + r][16 + nlo] = (_Float16)y1;
      }
    }
  }

  __syncthreads();

  // ---- Pooling (mean over tokens) + final LN, wave 0 only ----
  if (wave == 0) {
    float p = 0.0f;
#pragma unroll 8
    for (int t = 0; t < SS; ++t) p += xf[t][lane];
    p *= (1.0f / SS);
    float su = p, sq = p * p;
#pragma unroll
    for (int msk = 1; msk <= 16; msk <<= 1) {
      su += __shfl_xor(su, msk, 32);
      sq += __shfl_xor(sq, msk, 32);
    }
    float mean = su * (1.0f / DD);
    float var  = sq * (1.0f / DD) - mean * mean;
    float rstd = rsqrtf(var + EPS_);
    out[seq * DD + lane] = (p - mean) * rstd * lnf_g[lane] + lnf_b[lane];
  }
}

extern "C" void kernel_launch(void* const* d_in, const int* in_sizes, int n_in,
                              void* d_out, int out_size, void* d_ws, size_t ws_size,
                              hipStream_t stream) {
  (void)n_in; (void)out_size; (void)ws_size;
  const int*   token_ids = (const int*)d_in[0];
  const float* embed_W   = (const float*)d_in[1];
  const float* wqkv      = (const float*)d_in[2];
  const float* bqkv      = (const float*)d_in[3];
  const float* wo        = (const float*)d_in[4];
  const float* bo        = (const float*)d_in[5];
  const float* ln1_g     = (const float*)d_in[6];
  const float* ln1_b     = (const float*)d_in[7];
  const float* ln2_g     = (const float*)d_in[8];
  const float* ln2_b     = (const float*)d_in[9];
  const float* w1        = (const float*)d_in[10];
  const float* b1        = (const float*)d_in[11];
  const float* w2        = (const float*)d_in[12];
  const float* b2        = (const float*)d_in[13];
  const float* lnf_g     = (const float*)d_in[14];
  const float* lnf_b     = (const float*)d_in[15];

  float*    pe  = (float*)d_ws;
  _Float16* w16 = (_Float16*)(pe + SS * DD);

  sancta_prep_kernel<<<72, 256, 0, stream>>>(wqkv, wo, w1, w2, pe, w16);

  const int B = in_sizes[0] / SS;   // 8192 sequences
  sancta_tfm_kernel<<<B, 128, 0, stream>>>(token_ids, embed_W, bqkv, bo,
                                           ln1_g, ln1_b, ln2_g, ln2_b,
                                           b1, b2, lnf_g, lnf_b,
                                           pe, w16, (float*)d_out);
}
